// ClipLoss_hyperbolic_43430709297481
// MI455X (gfx1250) — matrix operand order repaired and checked by reading, not compiled, not run on hardware
//
#include <hip/hip_runtime.h>
#include <math.h>

#define BTOT 4096
#define DDIM 512
#define BPAD 40   // LDS row pitch in ushorts: 80B stride -> 16B aligned, bank-conflict-free

typedef __bf16 bf16x16 __attribute__((ext_vector_type(16)));
typedef float  f32x8   __attribute__((ext_vector_type(8)));

union FragU {
    bf16x16 v;
    uint4   q[2];
};

static __device__ inline unsigned short f2bf(float f) {
    unsigned u = __float_as_uint(f);
    unsigned r = (u + 0x7FFFu + ((u >> 16) & 1u)) >> 16;
    return (unsigned short)r;
}
static __device__ inline float bf2f(unsigned short h) {
    return __uint_as_float(((unsigned)h) << 16);
}

// Async-copy one 128x32-ushort tile (row stride DDIM in global) into LDS with
// row pitch BPAD. 512 16-byte chunks, 2 per thread. Tracked by ASYNCcnt.
static __device__ inline void stage_tile_async(const unsigned short* __restrict__ g0,
                                               unsigned lbase, int tid)
{
#pragma unroll
    for (int it = 0; it < 2; ++it) {
        int c   = tid + it * 256;       // chunk 0..511
        int row = c >> 2;               // 0..127
        int off = (c & 3) * 8;          // ushort offset within 32-wide row
        const unsigned short* g = g0 + (size_t)row * DDIM + off;
        unsigned l = lbase + (unsigned)(row * BPAD + off) * 2u;
        asm volatile("global_load_async_to_lds_b128 %0, %1, off"
                     :: "v"(l), "v"(g) : "memory");
    }
}

// ---------------- prep: bf16 hi/lo split + time components ----------------
__global__ __launch_bounds__(128)
void prep_kernel(const float* __restrict__ f0, const float* __restrict__ f1,
                 const float* __restrict__ f2,
                 unsigned short* __restrict__ hi, unsigned short* __restrict__ lo,
                 float* __restrict__ tvals, const float* __restrict__ curv_p)
{
    __shared__ float red[128];
    int r = blockIdx.x;                 // 0 .. 3*4096-1
    int f = r >> 12;
    int n = r & (BTOT - 1);
    const float* src = (f == 0 ? f0 : (f == 1 ? f1 : f2)) + (size_t)n * DDIM;
    unsigned short* hrow = hi + (size_t)r * DDIM;
    unsigned short* lrow = lo + (size_t)r * DDIM;

    float ss = 0.f;
    for (int d = threadIdx.x; d < DDIM; d += 128) {
        float x = src[d];
        unsigned short hb = f2bf(x);
        float hf = bf2f(hb);
        unsigned short lb = f2bf(x - hf);
        hrow[d] = hb;
        lrow[d] = lb;
        ss += x * x;
    }
    red[threadIdx.x] = ss;
    __syncthreads();
    for (int s = 64; s > 0; s >>= 1) {
        if (threadIdx.x < s) red[threadIdx.x] += red[threadIdx.x + s];
        __syncthreads();
    }
    if (threadIdx.x == 0)
        tvals[r] = sqrtf(1.0f / (*curv_p) + red[0]);
}

// ---------------- label match counts -------------------------------------
__global__ __launch_bounds__(256)
void cnt_kernel(const long long* __restrict__ labels, float* __restrict__ cnt)
{
    __shared__ long long lab[BTOT];     // 32 KB LDS
    for (int i = threadIdx.x; i < BTOT; i += 256) lab[i] = labels[i];
    __syncthreads();
    int n = blockIdx.x * 256 + threadIdx.x;
    long long mine = lab[n];
    int c = 0;
    for (int m = 0; m < BTOT; ++m) c += (lab[m] == mine) ? 1 : 0;
    cnt[n] = (float)c;
}

// ---------------- zero accumulators --------------------------------------
__global__ __launch_bounds__(256)
void zero_kernel(float* __restrict__ p, int n)
{
    int i = blockIdx.x * 256 + threadIdx.x;
    if (i < n) p[i] = 0.f;
}

// ---------------- fused bf16x3-WMMA GEMM + CE statistics ------------------
// grid (32, 32, 3): z picks the pair {(img,dna),(img,txt),(dna,txt)}.
// 256 threads = 8 waves (wave32). Wave w owns rows [M0+16w, M0+16w+16) x all
// 128 cols: A fragments are UNIQUE per wave (direct global loads, L2-resident)
// while the x8-shared B tile is staged into LDS once per block per K-step via
// global_load_async_to_lds_b128 (ASYNCcnt), double-buffered.
// Term-major WMMA sweeps (hi*hi, hi*lo, lo*hi) keep accumulator reuse distance
// at 8 WMMAs -> no bf16 WMMA->WMMA RAW hazard NOPs.
__global__ __launch_bounds__(256)
void gemm_ce_kernel(const unsigned short* __restrict__ hi,
                    const unsigned short* __restrict__ lo,
                    const float* __restrict__ tvals,
                    const long long* __restrict__ labels,
                    const float* __restrict__ logit_scale_p,
                    const float* __restrict__ curv_p,
                    float* __restrict__ rowSum, float* __restrict__ colSum,
                    float* __restrict__ Sacc)
{
    const int p  = blockIdx.z;
    const int ia = (p == 2) ? 1 : 0;
    const int ib = (p == 0) ? 1 : 2;
    const int M0 = blockIdx.x * 128;
    const int N0 = blockIdx.y * 128;

    const unsigned short* Ah = hi + (size_t)ia * BTOT * DDIM;
    const unsigned short* Al = lo + (size_t)ia * BTOT * DDIM;
    const unsigned short* Bh = hi + (size_t)ib * BTOT * DDIM;
    const unsigned short* Bl = lo + (size_t)ib * BTOT * DDIM;

    const int tid  = threadIdx.x;
    const int wave = tid >> 5;        // 0..7 : 16 rows each
    const int lane = tid & 31;
    const int lm   = lane & 15;
    const int h    = lane >> 4;

    __shared__ alignas(16) unsigned short BsH[2][128 * BPAD];   // 2 x 10 KB
    __shared__ alignas(16) unsigned short BsL[2][128 * BPAD];   // 2 x 10 KB
    __shared__ float ta[128], tb[128], rowPart[128], colPart[128];
    __shared__ long long la[128], lb[128];
    __shared__ float sPart;

    if (tid < 128) {
        ta[tid] = tvals[ia * BTOT + M0 + tid];
        tb[tid] = tvals[ib * BTOT + N0 + tid];
        la[tid] = labels[M0 + tid];
        lb[tid] = labels[N0 + tid];
        rowPart[tid] = 0.f;
        colPart[tid] = 0.f;
    }
    if (tid == 0) sPart = 0.f;

    const unsigned short* gBh = Bh + (size_t)N0 * DDIM;
    const unsigned short* gBl = Bl + (size_t)N0 * DDIM;
    const unsigned lbH0 = (unsigned)(size_t)&BsH[0][0];
    const unsigned lbH1 = (unsigned)(size_t)&BsH[1][0];
    const unsigned lbL0 = (unsigned)(size_t)&BsL[0][0];
    const unsigned lbL1 = (unsigned)(size_t)&BsL[1][0];

    // prologue: stage K-block 0 into buffer 0
    stage_tile_async(gBh, lbH0, tid);
    stage_tile_async(gBl, lbL0, tid);
    asm volatile("s_wait_asynccnt 0x0" ::: "memory");
    __syncthreads();

    f32x8 acc[8] = {};
    const int rA = M0 + wave * 16 + lm;
    int buf = 0;

#pragma unroll 1
    for (int k0 = 0; k0 < DDIM; k0 += 32) {
        // stage next K-block into the other buffer while computing this one
        if (k0 + 32 < DDIM) {
            stage_tile_async(gBh + k0 + 32, (buf == 0) ? lbH1 : lbH0, tid);
            stage_tile_async(gBl + k0 + 32, (buf == 0) ? lbL1 : lbL0, tid);
        }

        // A fragments: unique per wave, direct from global (L2-resident)
        FragU AHf, ALf;
        {
            size_t abase = (size_t)rA * DDIM + k0;
            AHf.q[0] = *(const uint4*)(Ah + abase + h * 8);
            AHf.q[1] = *(const uint4*)(Ah + abase + 16 + h * 8);
            ALf.q[0] = *(const uint4*)(Al + abase + h * 8);
            ALf.q[1] = *(const uint4*)(Al + abase + 16 + h * 8);
        }
        const unsigned short* sBH = &BsH[buf][0];
        const unsigned short* sBL = &BsL[buf][0];

        // term-major sweeps: accumulator reuse distance = 8 WMMAs
#pragma unroll
        for (int ni = 0; ni < 8; ++ni) {
            FragU Bf;
            int br = (ni * 16 + lm) * BPAD + h * 8;
            Bf.q[0] = *(const uint4*)(sBH + br);
            Bf.q[1] = *(const uint4*)(sBH + br + 16);
            acc[ni] = __builtin_amdgcn_wmma_f32_16x16x32_bf16(
                false, AHf.v, false, Bf.v, (short)0, acc[ni], false, false);
        }
#pragma unroll
        for (int ni = 0; ni < 8; ++ni) {
            FragU Bf;
            int br = (ni * 16 + lm) * BPAD + h * 8;
            Bf.q[0] = *(const uint4*)(sBL + br);
            Bf.q[1] = *(const uint4*)(sBL + br + 16);
            acc[ni] = __builtin_amdgcn_wmma_f32_16x16x32_bf16(
                false, AHf.v, false, Bf.v, (short)0, acc[ni], false, false);
        }
#pragma unroll
        for (int ni = 0; ni < 8; ++ni) {
            FragU Bf;
            int br = (ni * 16 + lm) * BPAD + h * 8;
            Bf.q[0] = *(const uint4*)(sBH + br);
            Bf.q[1] = *(const uint4*)(sBH + br + 16);
            acc[ni] = __builtin_amdgcn_wmma_f32_16x16x32_bf16(
                false, ALf.v, false, Bf.v, (short)0, acc[ni], false, false);
        }

        asm volatile("s_wait_asynccnt 0x0" ::: "memory");
        __syncthreads();
        buf ^= 1;
    }

    // ---- fused hyperbolic-distance + softmax-statistics epilogue ----
    // Row index rl invariant across ni; col index cl invariant across v.
    const float scale  = *logit_scale_p;
    const float curv   = *curv_p;
    const float factor = scale / sqrtf(curv);
    float myS = 0.f;
    float colAcc[8] = {0.f, 0.f, 0.f, 0.f, 0.f, 0.f, 0.f, 0.f};

#pragma unroll
    for (int v = 0; v < 8; ++v) {
        // C/D layout: lanes 0-15 -> M=v, lanes 16-31 -> M=v+8; N = lane%16
        int rl = wave * 16 + v + 8 * h;
        float tav = ta[rl];
        long long lav = la[rl];
        float rAcc = 0.f;
#pragma unroll
        for (int ni = 0; ni < 8; ++ni) {
            int cl = ni * 16 + lm;
            float inner = acc[ni][v];
            float c  = curv * (tav * tb[cl] - inner);
            float cc = fmaxf(c, 1.0f + 1e-8f);
            float dist = logf(cc + sqrtf(fmaxf(cc * cc - 1.0f, 0.0f)));
            float L = -factor * dist;   // logits; ~[-75,-60] -> exp stays normal fp32
            float e = __expf(L);
            rAcc += e;
            colAcc[ni] += e;
            if (lav == lb[cl]) myS += L;
        }
        atomicAdd(&rowPart[rl], rAcc);
    }
#pragma unroll
    for (int ni = 0; ni < 8; ++ni)
        atomicAdd(&colPart[ni * 16 + lm], colAcc[ni]);
    atomicAdd(&sPart, myS);
    __syncthreads();

    if (tid < 128) {
        atomicAdd(&rowSum[(size_t)p * BTOT + M0 + tid], rowPart[tid]);
        atomicAdd(&colSum[(size_t)p * BTOT + N0 + tid], colPart[tid]);
    }
    if (tid == 0) atomicAdd(&Sacc[p], sPart);
}

// ---------------- entailment loss (dna -> image) --------------------------
__global__ __launch_bounds__(128)
void entail_kernel(const float* __restrict__ img, const float* __restrict__ dna,
                   const float* __restrict__ tvals, const float* __restrict__ curv_p,
                   float* __restrict__ entailAcc)
{
    __shared__ float red[128];
    int n = blockIdx.x;
    const float* x = dna + (size_t)n * DDIM;   // fa = dna
    const float* y = img + (size_t)n * DDIM;   // fb = image
    float dot = 0.f;
    for (int d = threadIdx.x; d < DDIM; d += 128) dot += x[d] * y[d];
    red[threadIdx.x] = dot;
    __syncthreads();
    for (int s = 64; s > 0; s >>= 1) {
        if (threadIdx.x < s) red[threadIdx.x] += red[threadIdx.x + s];
        __syncthreads();
    }
    if (threadIdx.x == 0) {
        float curv = *curv_p;
        float xt = tvals[1 * BTOT + n];
        float yt = tvals[0 * BTOT + n];
        float nx = sqrtf(fmaxf(xt * xt - 1.0f / curv, 0.f));   // ||x||
        float cx = curv * (red[0] - xt * yt);
        float num = yt + cx * xt;
        float den = nx * sqrtf(fmaxf(cx * cx - 1.0f, 0.0f));
        float ai  = num / (den + 1e-8f);
        ai = fminf(fmaxf(ai, -1.0f + 1e-8f), 1.0f - 1e-8f);
        float ang = acosf(ai);
        float as_in = 0.2f / (nx * sqrtf(curv) + 1e-6f);
        as_in = fminf(fmaxf(as_in, -1.0f + 1e-6f), 1.0f - 1e-6f);
        float ap = asinf(as_in);
        atomicAdd(entailAcc, fmaxf(ang - ap, 0.0f));
    }
}

// ---------------- finalize ------------------------------------------------
__global__ __launch_bounds__(256)
void finalize_kernel(const float* __restrict__ rowSum, const float* __restrict__ colSum,
                     const float* __restrict__ Sacc, const float* __restrict__ cnt,
                     const float* __restrict__ entailAcc, float* __restrict__ out)
{
    __shared__ float red[256];
    int tid = threadIdx.x;
    float accum = 0.f;
    for (int p = 0; p < 3; ++p)
        for (int n = tid; n < BTOT; n += 256)
            accum += cnt[n] * (logf(rowSum[p * BTOT + n]) + logf(colSum[p * BTOT + n]));
    red[tid] = accum;
    __syncthreads();
    for (int s = 128; s > 0; s >>= 1) {
        if (tid < s) red[tid] += red[tid + s];
        __syncthreads();
    }
    if (tid == 0) {
        float Ssum = Sacc[0] + Sacc[1] + Sacc[2];
        float contrastive = (red[0] - 2.0f * Ssum) / (6.0f * (float)BTOT);
        float entail = entailAcc[0] / (float)BTOT;
        out[0] = contrastive + 0.2f * entail;   // total
        out[1] = contrastive;                    // contrastive_total
        out[2] = entail;                         // entail_total
    }
}

// ---------------- workspace layout (bytes) --------------------------------
static const size_t HI_OFF  = 0;                                   // 3*4096*512 bf16
static const size_t LO_OFF  = (size_t)3 * BTOT * DDIM * 2;         // 12,582,912
static const size_t T_OFF   = LO_OFF * 2;                          // 25,165,824
static const size_t CNT_OFF = T_OFF + (size_t)3 * BTOT * 4;        // 25,214,976
static const size_t ROW_OFF = CNT_OFF + (size_t)BTOT * 4;          // 25,231,360
static const size_t COL_OFF = ROW_OFF + (size_t)3 * BTOT * 4;      // 25,280,512
static const size_t S_OFF   = COL_OFF + (size_t)3 * BTOT * 4;      // 25,329,664
static const size_t ENT_OFF = S_OFF + 12;                          // 25,329,676
// zero region: ROW_OFF .. ENT_OFF+4  ->  3*4096*2 + 3 + 1 = 24580 floats

extern "C" void kernel_launch(void* const* d_in, const int* in_sizes, int n_in,
                              void* d_out, int out_size, void* d_ws, size_t ws_size,
                              hipStream_t stream) {
    (void)in_sizes; (void)n_in; (void)out_size; (void)ws_size;
    const float*     img    = (const float*)d_in[0];
    const float*     dna    = (const float*)d_in[1];
    const float*     txt    = (const float*)d_in[2];
    const long long* labels = (const long long*)d_in[3];
    const float*     lsp    = (const float*)d_in[4];
    const float*     cvp    = (const float*)d_in[5];
    float* out = (float*)d_out;

    char* ws = (char*)d_ws;
    unsigned short* hi   = (unsigned short*)(ws + HI_OFF);
    unsigned short* lo   = (unsigned short*)(ws + LO_OFF);
    float* tvals  = (float*)(ws + T_OFF);
    float* cnt    = (float*)(ws + CNT_OFF);
    float* rowSum = (float*)(ws + ROW_OFF);
    float* colSum = (float*)(ws + COL_OFF);
    float* Sacc   = (float*)(ws + S_OFF);
    float* entail = (float*)(ws + ENT_OFF);

    prep_kernel<<<dim3(3 * BTOT), 128, 0, stream>>>(img, dna, txt, hi, lo, tvals, cvp);
    cnt_kernel<<<dim3(BTOT / 256), 256, 0, stream>>>(labels, cnt);
    zero_kernel<<<dim3((24580 + 255) / 256), 256, 0, stream>>>(rowSum, 24580);
    gemm_ce_kernel<<<dim3(32, 32, 3), 256, 0, stream>>>(hi, lo, tvals, labels, lsp, cvp,
                                                        rowSum, colSum, Sacc);
    entail_kernel<<<dim3(BTOT), 128, 0, stream>>>(img, dna, tvals, cvp, entail);
    finalize_kernel<<<1, 256, 0, stream>>>(rowSum, colSum, Sacc, cnt, entail, out);
}
